// CrossAttention_20804821582346
// MI455X (gfx1250) — matrix operand order repaired and checked by reading, not compiled, not run on hardware
//
#include <hip/hip_runtime.h>
#include <hip/hip_bf16.h>

typedef __attribute__((ext_vector_type(16))) _Float16 v16h;
typedef __attribute__((ext_vector_type(8)))  _Float16 v8h;
typedef __attribute__((ext_vector_type(8)))  float    v8f;
typedef __attribute__((ext_vector_type(4)))  int      v4i;

#define AS1 __attribute__((address_space(1)))
#define AS3 __attribute__((address_space(3)))

namespace {
constexpr int kB  = 4;
constexpr int kV  = 6;
constexpr int kC  = 256;
constexpr int kHW = 1024;     // H*W
constexpr int kS  = kV * kHW; // 6144
constexpr int kVC = kV * kC;  // 1536
constexpr int kNH = 4;
constexpr int kHD = 64;
constexpr int kWIN = 3;
}

// ---------- helpers ----------
__device__ __forceinline__ v16h ld_a_pair(const _Float16* p) {
  // A-operand per-lane layout: 16 contiguous halfs at p (K 16h..16h+15)
  v8h lo = *(const v8h*)(p);
  v8h hi = *(const v8h*)(p + 8);
  v16h r;
#pragma unroll
  for (int i = 0; i < 8; ++i) { r[i] = lo[i]; r[i + 8] = hi[i]; }
  return r;
}

__device__ __forceinline__ v8f wmma_f16f32(v16h a, v16h b, v8f c) {
  return __builtin_amdgcn_wmma_f32_16x16x32_f16(false, a, false, b, (short)0, c,
                                                false, false);
}

// async 16-byte global -> LDS copy (ASYNCcnt-tracked, no VGPR staging)
__device__ __forceinline__ void async_copy16(const void* g, void* l) {
  __builtin_amdgcn_global_load_async_to_lds_b128((AS1 v4i*)g, (AS3 v4i*)l, 0, 0);
}

// ---------- kernel 1: pack x into f16 q-layout and kv-layout ----------
__global__ void pack_x_kernel(const float* __restrict__ x,
                              _Float16* __restrict__ q_in,   // [B, S, C]
                              _Float16* __restrict__ kv_in)  // [B, HW, V*C]
{
  int idx = blockIdx.x * blockDim.x + threadIdx.x;
  int total = kB * kV * kC * kHW;
  if (idx >= total) return;
  int hw = idx % kHW;
  int t  = idx / kHW;
  int c  = t % kC; t /= kC;
  int v  = t % kV;
  int b  = t / kV;
  _Float16 val = (_Float16)x[idx];  // x is [b][v][c][hw] with hw fastest == idx
  q_in[((size_t)b * kS + (size_t)v * kHW + hw) * kC + c] = val;
  kv_in[((size_t)b * kHW + hw) * kVC + (size_t)v * kC + c] = val;
}

// ---------- kernel 2: transpose+convert weight [K,N] f32 -> [N,K] f16 ----------
__global__ void pack_w_kernel(const float* __restrict__ w, _Float16* __restrict__ wt,
                              int K, int N)
{
  int idx = blockIdx.x * blockDim.x + threadIdx.x;
  if (idx >= K * N) return;
  int n = idx % N;
  int k = idx / N;
  wt[(size_t)n * K + k] = (_Float16)w[idx];
}

// ---------- kernel 3: WMMA GEMM, f16 out  D = A[MxK] * Wt^T + bias ----------
// Wt is stored [N, K] (pre-transposed).  mode 0: out row-major [M,N] f16.
// mode 1: out is V^T per head: [B=M/1024][nh=4][hd=64][HW=1024] f16.
__global__ void gemm_f16_kernel(const _Float16* __restrict__ A,
                                const _Float16* __restrict__ Wt,
                                const float* __restrict__ bias,
                                int M, int N, int K,
                                _Float16* __restrict__ out, int mode)
{
  int lane = threadIdx.x & 31;
  int wid  = (blockIdx.x * blockDim.x + threadIdx.x) >> 5;
  int ntiles = N / 64;
  int ttot = (M / 16) * ntiles;
  if (wid >= ttot) return;
  int mtile = wid / ntiles;
  int nt    = wid % ntiles;
  int m0 = mtile * 16, n0 = nt * 64;

  int arow = lane & 15, ach = (lane >> 4) * 16;
  int bcol = lane & 15, bko = (lane >> 4) * 16;
  const _Float16* Ap = A + (size_t)(m0 + arow) * K + ach;

  v8f acc[4] = {};
  for (int k0 = 0; k0 < K; k0 += 32) {
    __builtin_prefetch(Ap + k0 + 128, 0, 0);   // pull next A block toward L1/L2
    v16h a = ld_a_pair(Ap + k0);
#pragma unroll
    for (int t = 0; t < 4; ++t) {
      const _Float16* Bp = Wt + (size_t)(n0 + t * 16 + bcol) * K + k0 + bko;
      v16h bop = *(const v16h*)Bp;
      acc[t] = wmma_f16f32(a, bop, acc[t]);
    }
  }

  int cn = lane & 15, mb = (lane >> 4) * 8;
#pragma unroll
  for (int t = 0; t < 4; ++t) {
    int n = n0 + t * 16 + cn;
    float bv = bias[n];
#pragma unroll
    for (int r = 0; r < 8; ++r) {
      float val = acc[t][r] + bv;
      int m = m0 + mb + r;
      if (mode == 0) {
        out[(size_t)m * N + n] = (_Float16)val;
      } else {
        // m = b*1024 + hw ; n = h*64 + d  ->  vT[((b*4+h)*64+d)*1024 + hw]
        int b = m >> 10, hw = m & 1023;
        int h = n >> 6,  d  = n & 63;
        out[((((size_t)b * kNH + h) * kHD) + d) * kHW + hw] = (_Float16)val;
      }
    }
  }
}

// ---------- kernel 4: flash attention ----------
// One wave per (b, h, 16 query rows); the 4 waves of a block share (b, h) and
// cooperatively stage each 32-wide K/V chunk into LDS with async b128 copies.
__global__ void __launch_bounds__(128)
attn_kernel(const _Float16* __restrict__ qh,  // [B,S,C] f16
            const _Float16* __restrict__ kh,  // [B,HW,C] f16
            const _Float16* __restrict__ vT,  // [B,nh,hd,HW] f16
            _Float16* __restrict__ oh)        // [B,S,C] f16
{
  __shared__ __align__(32) _Float16 kbuf[32 * 64];   // [j in chunk][d]   4 KB
  __shared__ __align__(32) _Float16 vbuf[64 * 32];   // [d][j in chunk]   4 KB
  __shared__ __align__(32) _Float16 plds[4][16 * 32];//                   4 KB
  int tid  = threadIdx.x;
  int lane = tid & 31;
  int wloc = tid >> 5;
  int wid  = blockIdx.x * 4 + wloc;
  int nqt  = kS / 16; // 384
  int qt = wid % nqt;
  int t0 = wid / nqt;
  int h  = t0 % kNH;
  int b  = t0 / kNH;
  int q0 = qt * 16;

  int arow = lane & 15, ach = (lane >> 4) * 16;
  int bcol = lane & 15, bko = (lane >> 4) * 16;
  int cn   = lane & 15, mb  = (lane >> 4) * 8;

  // Q A-operands for the two K-steps of hd=64
  const _Float16* Qp = qh + ((size_t)b * kS + q0 + arow) * kC + h * kHD + ach;
  v16h qa0 = ld_a_pair(Qp);
  v16h qa1 = ld_a_pair(Qp + 32);

  v8f o[4] = {};
  float rm[8], rl[8];
#pragma unroll
  for (int r = 0; r < 8; ++r) { rm[r] = -3.0e38f; rl[r] = 0.0f; }

  for (int j0 = 0; j0 < kHW; j0 += 32) {
    // ---- cooperative async stage of K (32x64) and V^T (64x32) chunks ----
#pragma unroll
    for (int it = 0; it < 2; ++it) {           // K chunk: 256 x 16B
      int e   = tid + it * 128;
      int row = e >> 3;                        // j within chunk, 0..31
      int ho  = (e & 7) * 8;                   // half offset in row
      const _Float16* g = kh + ((size_t)b * kHW + j0 + row) * kC + h * kHD + ho;
      async_copy16(g, (char*)kbuf + (size_t)e * 16);
    }
#pragma unroll
    for (int it = 0; it < 2; ++it) {           // V chunk: 256 x 16B
      int e   = tid + it * 128;
      int row = e >> 2;                        // d, 0..63
      int ho  = (e & 3) * 8;
      const _Float16* g = vT + (((size_t)b * kNH + h) * kHD + row) * kHW + j0 + ho;
      async_copy16(g, (char*)vbuf + (size_t)e * 16);
    }
    asm volatile("s_wait_asynccnt 0x0" ::: "memory");
    __syncthreads();

    // ---- logits: S = Q(16x64) @ K^T(64x32), operands from LDS ----
    v8f s[2] = {};
#pragma unroll
    for (int ns = 0; ns < 2; ++ns) {
      const _Float16* Kp = &kbuf[(ns * 16 + bcol) * 64 + bko];
      v16h kb0 = *(const v16h*)(Kp);
      v16h kb1 = *(const v16h*)(Kp + 32);
      s[ns] = wmma_f16f32(qa0, kb0, s[ns]);
      s[ns] = wmma_f16f32(qa1, kb1, s[ns]);
    }
    // scale + additive window mask + per-row max over this 32-col chunk
    float tmax[8];
#pragma unroll
    for (int r = 0; r < 8; ++r) tmax[r] = -3.0e38f;
#pragma unroll
    for (int ns = 0; ns < 2; ++ns) {
      int j = j0 + ns * 16 + cn;
#pragma unroll
      for (int r = 0; r < 8; ++r) {
        float lg = s[ns][r] * 0.125f;          // 1/sqrt(64)
        int i = q0 + mb + r;
        if (j >= i - kWIN && j <= i + kWIN) lg -= 1.0e9f;  // mask OUT window
        s[ns][r] = lg;
        tmax[r] = fmaxf(tmax[r], lg);
      }
    }
#pragma unroll
    for (int msk = 1; msk <= 8; msk <<= 1)
#pragma unroll
      for (int r = 0; r < 8; ++r)
        tmax[r] = fmaxf(tmax[r], __shfl_xor(tmax[r], msk, 32));

    float alpha[8], rsum[8];
#pragma unroll
    for (int r = 0; r < 8; ++r) {
      float mnew = fmaxf(rm[r], tmax[r]);
      alpha[r] = __expf(rm[r] - mnew);
      rm[r] = mnew;
      rsum[r] = 0.0f;
    }
    // P = exp(s - m) -> per-wave LDS (row-major 16x32 f16) for C/D -> A relayout
#pragma unroll
    for (int ns = 0; ns < 2; ++ns) {
#pragma unroll
      for (int r = 0; r < 8; ++r) {
        float p = __expf(s[ns][r] - rm[r]);
        rsum[r] += p;
        plds[wloc][(mb + r) * 32 + ns * 16 + cn] = (_Float16)p;
      }
    }
#pragma unroll
    for (int msk = 1; msk <= 8; msk <<= 1)
#pragma unroll
      for (int r = 0; r < 8; ++r) rsum[r] += __shfl_xor(rsum[r], msk, 32);
#pragma unroll
    for (int r = 0; r < 8; ++r) rl[r] = rl[r] * alpha[r] + rsum[r];
#pragma unroll
    for (int t = 0; t < 4; ++t)
#pragma unroll
      for (int r = 0; r < 8; ++r) o[t][r] *= alpha[r];

    asm volatile("s_wait_dscnt 0x0" ::: "memory");  // same-wave LDS RAW fence
    v16h pa = ld_a_pair(&plds[wloc][arow * 32 + ach]);

    // ---- O += P(16x32) @ V(32x64), B operand from staged V^T in LDS ----
#pragma unroll
    for (int t = 0; t < 4; ++t) {
      const _Float16* Vp = &vbuf[(t * 16 + bcol) * 32 + bko];
      v16h vb = *(const v16h*)Vp;
      o[t] = wmma_f16f32(pa, vb, o[t]);
    }
    __syncthreads();   // protect kbuf/vbuf before next chunk's async stores
  }

  // normalize and store f16 [B,S,C]
#pragma unroll
  for (int t = 0; t < 4; ++t) {
#pragma unroll
    for (int r = 0; r < 8; ++r) {
      float val = o[t][r] / rl[r];
      int m = q0 + mb + r;
      int n = h * kHD + t * 16 + cn;
      oh[((size_t)b * kS + m) * kC + n] = (_Float16)val;
    }
  }
}

// ---------- kernel 5: output projection + bias + residual, f32 scatter ----------
__global__ void gemm_out_kernel(const _Float16* __restrict__ A,   // [B*S, C] f16
                                const _Float16* __restrict__ WoT, // [C, C] f16 (N,K)
                                const float* __restrict__ bo,
                                const float* __restrict__ x,      // [B*V, C, HW] f32
                                float* __restrict__ out)          // same layout
{
  const int M = kB * kS, N = kC, K = kC;
  int lane = threadIdx.x & 31;
  int wid  = (blockIdx.x * blockDim.x + threadIdx.x) >> 5;
  int ntiles = N / 64;
  if (wid >= (M / 16) * ntiles) return;
  int mtile = wid / ntiles, nt = wid % ntiles;
  int m0 = mtile * 16, n0 = nt * 64;

  int arow = lane & 15, ach = (lane >> 4) * 16;
  int bcol = lane & 15, bko = (lane >> 4) * 16;
  const _Float16* Ap = A + (size_t)(m0 + arow) * K + ach;

  v8f acc[4] = {};
  for (int k0 = 0; k0 < K; k0 += 32) {
    __builtin_prefetch(Ap + k0 + 128, 0, 0);
    v16h a = ld_a_pair(Ap + k0);
#pragma unroll
    for (int t = 0; t < 4; ++t) {
      const _Float16* Bp = WoT + (size_t)(n0 + t * 16 + bcol) * K + k0 + bko;
      v16h bop = *(const v16h*)Bp;
      acc[t] = wmma_f16f32(a, bop, acc[t]);
    }
  }

  int cn = lane & 15, mb = (lane >> 4) * 8;
#pragma unroll
  for (int t = 0; t < 4; ++t) {
    int n = n0 + t * 16 + cn;
    float bv = bo[n];
#pragma unroll
    for (int r = 0; r < 8; ++r) {
      int m = m0 + mb + r;              // m = b*S + s
      int b = m / kS, s = m % kS;
      int v = s >> 10, hw = s & 1023;
      size_t oi = (((size_t)b * kV + v) * kC + n) * kHW + hw;
      out[oi] = x[oi] + acc[t][r] + bv;
    }
  }
}

// ---------- launch ----------
extern "C" void kernel_launch(void* const* d_in, const int* in_sizes, int n_in,
                              void* d_out, int out_size, void* d_ws, size_t ws_size,
                              hipStream_t stream) {
  (void)in_sizes; (void)n_in; (void)out_size; (void)ws_size;
  const float* x  = (const float*)d_in[0];
  const float* Wq = (const float*)d_in[1];
  const float* bq = (const float*)d_in[2];
  const float* Wk = (const float*)d_in[3];
  const float* bk = (const float*)d_in[4];
  const float* Wv = (const float*)d_in[5];
  const float* bv = (const float*)d_in[6];
  const float* Wo = (const float*)d_in[7];
  const float* bo = (const float*)d_in[8];
  float* out = (float*)d_out;

  char* ws = (char*)d_ws;
  size_t off = 0;
  auto alloc = [&](size_t bytes) { char* p = ws + off; off += (bytes + 255) & ~(size_t)255; return p; };
  _Float16* q_in  = (_Float16*)alloc((size_t)kB * kS * kC * 2);
  _Float16* kv_in = (_Float16*)alloc((size_t)kB * kHW * kVC * 2);
  _Float16* WqT   = (_Float16*)alloc((size_t)kC * kC * 2);
  _Float16* WkT   = (_Float16*)alloc((size_t)kVC * kC * 2);
  _Float16* WvT   = (_Float16*)alloc((size_t)kVC * kC * 2);
  _Float16* WoT   = (_Float16*)alloc((size_t)kC * kC * 2);
  _Float16* qH    = (_Float16*)alloc((size_t)kB * kS * kC * 2);
  _Float16* kH    = (_Float16*)alloc((size_t)kB * kHW * kC * 2);
  _Float16* vTH   = (_Float16*)alloc((size_t)kB * kHW * kC * 2);
  _Float16* aH    = (_Float16*)alloc((size_t)kB * kS * kC * 2);

  // 1) pack activations + weights
  {
    int total = kB * kV * kC * kHW;
    pack_x_kernel<<<(total + 255) / 256, 256, 0, stream>>>(x, q_in, kv_in);
    pack_w_kernel<<<(kC * kC + 255) / 256, 256, 0, stream>>>(Wq, WqT, kC, kC);
    pack_w_kernel<<<(kVC * kC + 255) / 256, 256, 0, stream>>>(Wk, WkT, kVC, kC);
    pack_w_kernel<<<(kVC * kC + 255) / 256, 256, 0, stream>>>(Wv, WvT, kVC, kC);
    pack_w_kernel<<<(kC * kC + 255) / 256, 256, 0, stream>>>(Wo, WoT, kC, kC);
  }
  // 2) projections (WMMA)
  {
    int Mq = kB * kS;                         // 24576
    int waves = (Mq / 16) * (kC / 64);        // 6144
    gemm_f16_kernel<<<waves / 4, 128, 0, stream>>>(q_in, WqT, bq, Mq, kC, kC, qH, 0);
    int Mkv = kB * kHW;                       // 4096
    int wkv = (Mkv / 16) * (kC / 64);         // 1024
    gemm_f16_kernel<<<wkv / 4, 128, 0, stream>>>(kv_in, WkT, bk, Mkv, kC, kVC, kH, 0);
    gemm_f16_kernel<<<wkv / 4, 128, 0, stream>>>(kv_in, WvT, bv, Mkv, kC, kVC, vTH, 1);
  }
  // 3) attention (WMMA flash-style, async LDS staging)
  {
    int waves = kB * kNH * (kS / 16);         // 6144
    attn_kernel<<<waves / 4, 128, 0, stream>>>(qH, kH, vTH, aH);
  }
  // 4) output projection + residual (WMMA)
  {
    int waves = ((kB * kS) / 16) * (kC / 64); // 6144
    gemm_out_kernel<<<waves / 4, 128, 0, stream>>>(aH, WoT, bo, x, out);
  }
}
